// Transformer2DBlock_76390288327293
// MI455X (gfx1250) — compile-verified, matrix operand order checked by reading, not run on hardware
//
#include <hip/hip_runtime.h>
#include <hip/hip_bf16.h>
#include <math.h>

typedef _Float16 h16;
typedef __attribute__((ext_vector_type(16))) _Float16 v16h;
typedef __attribute__((ext_vector_type(8)))  _Float16 v8h;
typedef __attribute__((ext_vector_type(8)))  float    v8f;

#define NWIN   2048     // (B * LON/WW) * (LAT/WL) = 64*32
#define NPAD   80       // 72 tokens padded to multiple of 16
#define NTOK   72
#define TROWS  147456   // B * LAT * LON
#define CDIM   192
#define HID    768
#define QKVO   576

// ---------------- WMMA helpers ----------------

__device__ __forceinline__ v16h load_frag(const h16* base, int ld, int k0) {
  // A-frag: rows = lane&15, K split per documented CDNA5 16-bit layout.
  // Doubles as B-frag loader when "base" points at the matrix whose
  // *columns* (over K) are contiguous rows of length ld.
  int lane = threadIdx.x & 31;
  int r    = lane & 15;
  int koff = (lane >> 4) << 3;          // 0 or 8
  const h16* p = base + (size_t)r * ld + k0 + koff;
  v8h lo = *(const v8h*)p;
  v8h hi = *(const v8h*)(p + 16);
  v16h out;
#pragma unroll
  for (int i = 0; i < 8; ++i) { out[i] = lo[i]; out[i + 8] = hi[i]; }
  return out;
}

__device__ __forceinline__ v8f wmma_f16(v16h a, v16h b, v8f c) {
  return __builtin_amdgcn_wmma_f32_16x16x32_f16(
      /*neg_a=*/false, a, /*neg_b=*/false, b,
      /*c_mod=*/(short)0, c, /*reuse_a=*/false, /*reuse_b=*/false);
}

// 16 (rows) x 64 (cols) output patch per wave, register double-buffered over
// K so next-chunk loads issue while current-chunk WMMAs execute.
template <int K>
__device__ __forceinline__ void gemm16x64(const h16* A, int lda,
                                          const h16* B, int ldb,
                                          v8f acc[4]) {
  v16h a_cur = load_frag(A, lda, 0);
  v16h b_cur[4];
#pragma unroll
  for (int t = 0; t < 4; ++t) b_cur[t] = load_frag(B + (size_t)(t * 16) * ldb, ldb, 0);

#pragma unroll
  for (int k0 = 0; k0 < K; k0 += 32) {
    v16h a_nxt = a_cur;
    v16h b_nxt[4];
    if (k0 + 32 < K) {
      a_nxt = load_frag(A, lda, k0 + 32);
#pragma unroll
      for (int t = 0; t < 4; ++t)
        b_nxt[t] = load_frag(B + (size_t)(t * 16) * ldb, ldb, k0 + 32);
    }
#pragma unroll
    for (int t = 0; t < 4; ++t) acc[t] = wmma_f16(a_cur, b_cur[t], acc[t]);
    a_cur = a_nxt;
    if (k0 + 32 < K) {
#pragma unroll
      for (int t = 0; t < 4; ++t) b_cur[t] = b_nxt[t];
    }
  }
}

__device__ __forceinline__ float wave_sum(float v) {
#pragma unroll
  for (int off = 16; off; off >>= 1) v += __shfl_xor(v, off, 32);
  return v;
}
__device__ __forceinline__ float wave_max(float v) {
#pragma unroll
  for (int off = 16; off; off >>= 1) v = fmaxf(v, __shfl_xor(v, off, 32));
  return v;
}

// ---------------- Kernels ----------------

__global__ void k_f32_to_f16(const float* __restrict__ s, h16* __restrict__ d, int n) {
  int i = blockIdx.x * 256 + threadIdx.x;
  if (i < n) d[i] = (h16)s[i];
}

// LN1 + roll(-3,-6) + window partition -> hw f16 [NWIN][NPAD][192]
__global__ void k_ln1_part(const float* __restrict__ x,
                           const float* __restrict__ w, const float* __restrict__ b,
                           h16* __restrict__ hw) {
  int wave = threadIdx.x >> 5, lane = threadIdx.x & 31;
  long row = (long)blockIdx.x * 8 + wave;
  if (row >= (long)NWIN * NPAD) return;
  int  n   = (int)(row % NPAD);
  long win = row / NPAD;
  h16* dst = hw + (size_t)row * CDIM;
  if (n >= NTOK) {
#pragma unroll
    for (int i = 0; i < 6; ++i) dst[lane + 32 * i] = (h16)0.f;
    return;
  }
  int latb = (int)(win & 31), lonb = (int)((win >> 5) & 31), bb = (int)(win >> 10);
  int wl = n / 12, ww = n % 12;
  int sl = (latb * 6 + wl + 3) % 192;
  int so = (lonb * 12 + ww + 6) % 384;
  const float* src = x + ((size_t)bb * 73728 + (size_t)sl * 384 + so) * CDIM;
  float v[6]; float s = 0.f;
#pragma unroll
  for (int i = 0; i < 6; ++i) { v[i] = src[lane + 32 * i]; s += v[i]; }
  float mean = wave_sum(s) * (1.f / CDIM);
  float vs = 0.f;
#pragma unroll
  for (int i = 0; i < 6; ++i) { float d0 = v[i] - mean; vs += d0 * d0; }
  float var = wave_sum(vs) * (1.f / CDIM);
  float rsq = rsqrtf(var + 1e-5f);
#pragma unroll
  for (int i = 0; i < 6; ++i) {
    int c = lane + 32 * i;
    dst[c] = (h16)((v[i] - mean) * rsq * w[c] + b[c]);
  }
}

// QKV GEMM: hw [rows][192] x qkvW [576][192]^T -> qkv f16 [rows][576]
// q part (cols<192) pre-scaled by hd^-0.5 (after bias, matching reference).
__global__ __launch_bounds__(32) void k_qkv(const h16* __restrict__ hw,
                                            const h16* __restrict__ W,
                                            const float* __restrict__ bias,
                                            h16* __restrict__ qkv) {
  int mt = blockIdx.x, col0 = blockIdx.y * 64;
  v8f acc[4] = {};
  gemm16x64<CDIM>(hw + (size_t)mt * 16 * CDIM, CDIM, W + (size_t)col0 * CDIM, CDIM, acc);
  int lane = threadIdx.x & 31, rb = (lane >> 4) * 8, col = lane & 15;
  size_t row0 = (size_t)mt * 16;
#pragma unroll
  for (int t = 0; t < 4; ++t)
#pragma unroll
    for (int r = 0; r < 8; ++r) {
      int c = col0 + t * 16 + col;
      float v = acc[t][r] + bias[c];
      if (c < CDIM) v *= 0.1767766953f;    // 1/sqrt(32)
      qkv[(row0 + rb + r) * QKVO + c] = (h16)v;
    }
}

// Attention: one wave per (window, head). QK^T + earth bias + softmax + AV.
__global__ __launch_bounds__(32) void k_attn(const h16* __restrict__ qkv,
                                             const float* __restrict__ table,
                                             const int* __restrict__ epi,
                                             h16* __restrict__ obuf) {
  __shared__ __align__(16) float sc[NPAD * 96];
  __shared__ __align__(16) h16   pr[NPAD * 96];
  __shared__ __align__(16) h16   vT[32 * 96];
  int win = blockIdx.x / 6, head = blockIdx.x % 6;
  int lane = threadIdx.x;
  int latb = win & 31;
  const h16* qp = qkv + (size_t)win * NPAD * QKVO + head * 32;
  const h16* kp = qp + CDIM;
  const h16* vp = qp + 2 * CDIM;

  for (int i = lane; i < 32 * 96; i += 32) vT[i] = (h16)0.f;
  __syncthreads();
  for (int i = lane; i < NTOK * 32; i += 32) {
    int m = i >> 5, d = i & 31;
    vT[d * 96 + m] = vp[(size_t)m * QKVO + d];
  }

  v16h aq[5];
#pragma unroll
  for (int i = 0; i < 5; ++i) aq[i] = load_frag(qp + (size_t)(16 * i) * QKVO, QKVO, 0);

  int rb = (lane >> 4) * 8, col = lane & 15;
  v16h bk_cur = load_frag(kp, QKVO, 0);
  for (int j = 0; j < 5; ++j) {
    v16h bk_nxt = bk_cur;
    if (j + 1 < 5) bk_nxt = load_frag(kp + (size_t)(16 * (j + 1)) * QKVO, QKVO, 0);
#pragma unroll
    for (int i = 0; i < 5; ++i) {
      v8f acc = {};
      acc = wmma_f16(aq[i], bk_cur, acc);
#pragma unroll
      for (int r = 0; r < 8; ++r) {
        int row = 16 * i + rb + r;
        int cg  = 16 * j + col;
        if (cg < NTOK) {
          float bv = 0.f;
          if (row < NTOK)
            bv = table[((size_t)epi[row * NTOK + cg] * 32 + latb) * 6 + head];
          sc[row * 96 + cg] = acc[r] + bv;
        }
      }
    }
    bk_cur = bk_nxt;
  }
  __syncthreads();

  for (int n = 0; n < NPAD; ++n) {
    if (n >= NTOK) {
#pragma unroll
      for (int q = 0; q < 3; ++q) pr[n * 96 + lane + 32 * q] = (h16)0.f;
      continue;
    }
    float s[3];
#pragma unroll
    for (int q = 0; q < 3; ++q) {
      int c = lane + 32 * q;
      s[q] = (c < NTOK) ? sc[n * 96 + c] : -1e30f;
    }
    float m = wave_max(fmaxf(fmaxf(s[0], s[1]), s[2]));
    float e[3], sum = 0.f;
#pragma unroll
    for (int q = 0; q < 3; ++q) { e[q] = __expf(s[q] - m); sum += e[q]; }
    float inv = 1.f / wave_sum(sum);
#pragma unroll
    for (int q = 0; q < 3; ++q) pr[n * 96 + lane + 32 * q] = (h16)(e[q] * inv);
  }
  __syncthreads();

  for (int i = 0; i < 5; ++i) {
    v8f a0 = {}, a1 = {};
#pragma unroll
    for (int kc = 0; kc < 3; ++kc) {
      v16h ap = load_frag(pr + (size_t)(16 * i) * 96, 96, kc * 32);
      v16h b0 = load_frag(vT, 96, kc * 32);
      v16h b1 = load_frag(vT + 16 * 96, 96, kc * 32);
      a0 = wmma_f16(ap, b0, a0);
      a1 = wmma_f16(ap, b1, a1);
    }
#pragma unroll
    for (int r = 0; r < 8; ++r) {
      int row = 16 * i + rb + r;
      h16* op = obuf + ((size_t)win * NPAD + row) * CDIM + head * 32;
      op[col]      = (h16)a0[r];
      op[16 + col] = (h16)a1[r];
    }
  }
}

// Proj GEMM + window reverse + roll(+3,+6) + residual -> x_mid f32 [TROWS][192]
__global__ __launch_bounds__(32) void k_proj(const h16* __restrict__ obuf,
                                             const h16* __restrict__ W,
                                             const float* __restrict__ bias,
                                             const float* __restrict__ x,
                                             float* __restrict__ xmid) {
  int mt = blockIdx.x, col0 = blockIdx.y * 64;
  v8f acc[4] = {};
  gemm16x64<CDIM>(obuf + (size_t)mt * 16 * CDIM, CDIM, W + (size_t)col0 * CDIM, CDIM, acc);
  int lane = threadIdx.x & 31, rb = (lane >> 4) * 8, col = lane & 15;
#pragma unroll
  for (int r = 0; r < 8; ++r) {
    int grow = mt * 16 + rb + r;
    int n = grow % NPAD;
    if (n >= NTOK) continue;              // drop pad rows
    int win = grow / NPAD;
    int latb = win & 31, lonb = (win >> 5) & 31, bb = win >> 10;
    int wl = n / 12, ww = n % 12;
    int sl = (latb * 6 + wl + 3) % 192;
    int so = (lonb * 12 + ww + 6) % 384;
    size_t dst = ((size_t)bb * 73728 + (size_t)sl * 384 + so) * CDIM;
#pragma unroll
    for (int t = 0; t < 4; ++t) {
      int c = col0 + t * 16 + col;
      xmid[dst + c] = x[dst + c] + acc[t][r] + bias[c];
    }
  }
}

// LN2 -> h2 f16 [TROWS][192]
__global__ void k_ln2(const float* __restrict__ xmid,
                      const float* __restrict__ w, const float* __restrict__ b,
                      h16* __restrict__ h2) {
  int wave = threadIdx.x >> 5, lane = threadIdx.x & 31;
  long row = (long)blockIdx.x * 8 + wave;
  if (row >= (long)TROWS) return;
  const float* src = xmid + (size_t)row * CDIM;
  h16* dst = h2 + (size_t)row * CDIM;
  float v[6]; float s = 0.f;
#pragma unroll
  for (int i = 0; i < 6; ++i) { v[i] = src[lane + 32 * i]; s += v[i]; }
  float mean = wave_sum(s) * (1.f / CDIM);
  float vs = 0.f;
#pragma unroll
  for (int i = 0; i < 6; ++i) { float d0 = v[i] - mean; vs += d0 * d0; }
  float rsq = rsqrtf(wave_sum(vs) * (1.f / CDIM) + 1e-5f);
#pragma unroll
  for (int i = 0; i < 6; ++i) {
    int c = lane + 32 * i;
    dst[c] = (h16)((v[i] - mean) * rsq * w[c] + b[c]);
  }
}

// fc1 GEMM + exact GELU -> hid f16 [TROWS][768]
__global__ __launch_bounds__(32) void k_fc1(const h16* __restrict__ h2,
                                            const h16* __restrict__ W,
                                            const float* __restrict__ bias,
                                            h16* __restrict__ hid) {
  int mt = blockIdx.x, col0 = blockIdx.y * 64;
  v8f acc[4] = {};
  gemm16x64<CDIM>(h2 + (size_t)mt * 16 * CDIM, CDIM, W + (size_t)col0 * CDIM, CDIM, acc);
  int lane = threadIdx.x & 31, rb = (lane >> 4) * 8, col = lane & 15;
  size_t row0 = (size_t)mt * 16;
#pragma unroll
  for (int t = 0; t < 4; ++t)
#pragma unroll
    for (int r = 0; r < 8; ++r) {
      int c = col0 + t * 16 + col;
      float v = acc[t][r] + bias[c];
      float g = 0.5f * v * (1.f + erff(v * 0.70710678f));
      hid[(row0 + rb + r) * HID + c] = (h16)g;
    }
}

// fc2 GEMM + bias + residual -> out f32 [TROWS][192]
__global__ __launch_bounds__(32) void k_fc2(const h16* __restrict__ hid,
                                            const h16* __restrict__ W,
                                            const float* __restrict__ bias,
                                            const float* __restrict__ xmid,
                                            float* __restrict__ out) {
  int mt = blockIdx.x, col0 = blockIdx.y * 64;
  v8f acc[4] = {};
  gemm16x64<HID>(hid + (size_t)mt * 16 * HID, HID, W + (size_t)col0 * HID, HID, acc);
  int lane = threadIdx.x & 31, rb = (lane >> 4) * 8, col = lane & 15;
  size_t row0 = (size_t)mt * 16;
#pragma unroll
  for (int t = 0; t < 4; ++t)
#pragma unroll
    for (int r = 0; r < 8; ++r) {
      int c = col0 + t * 16 + col;
      size_t row = row0 + rb + r;
      out[row * CDIM + c] = xmid[row * CDIM + c] + acc[t][r] + bias[c];
    }
}

// ---------------- Launch ----------------

extern "C" void kernel_launch(void* const* d_in, const int* in_sizes, int n_in,
                              void* d_out, int out_size, void* d_ws, size_t ws_size,
                              hipStream_t stream) {
  (void)in_sizes; (void)n_in; (void)out_size; (void)ws_size;
  const float* x      = (const float*)d_in[0];
  const float* n1w    = (const float*)d_in[1];
  const float* n1b    = (const float*)d_in[2];
  const float* qkv_w  = (const float*)d_in[3];
  const float* qkv_b  = (const float*)d_in[4];
  const float* table  = (const float*)d_in[5];
  const float* proj_w = (const float*)d_in[6];
  const float* proj_b = (const float*)d_in[7];
  const float* n2w    = (const float*)d_in[8];
  const float* n2b    = (const float*)d_in[9];
  const float* fc1_w  = (const float*)d_in[10];
  const float* fc1_b  = (const float*)d_in[11];
  const float* fc2_w  = (const float*)d_in[12];
  const float* fc2_b  = (const float*)d_in[13];
  const int*   epi    = (const int*)d_in[14];

  char* base = (char*)d_ws;
  h16* qkvW = (h16*)base;                    // 576*192 halfs
  h16* projW = qkvW + 110592;                // 192*192
  h16* fc1W  = qkvW + 147456;                // 768*192
  h16* fc2W  = qkvW + 294912;                // 192*768  (ends at 442368 halfs)
  h16*  bufA = (h16*)(base + (1ull  << 20));   // 60 MB region: hw / obuf / h2
  h16*  bufB = (h16*)(base + (64ull << 20));   // 216 MB region: qkv / hid
  float* bufX = (float*)(base + (280ull << 20)); // 108 MB: x_mid

  k_f32_to_f16<<<(110592 + 255) / 256, 256, 0, stream>>>(qkv_w, qkvW, 110592);
  k_f32_to_f16<<<(36864  + 255) / 256, 256, 0, stream>>>(proj_w, projW, 36864);
  k_f32_to_f16<<<(147456 + 255) / 256, 256, 0, stream>>>(fc1_w, fc1W, 147456);
  k_f32_to_f16<<<(147456 + 255) / 256, 256, 0, stream>>>(fc2_w, fc2W, 147456);

  k_ln1_part<<<NWIN * NPAD / 8, 256, 0, stream>>>(x, n1w, n1b, bufA);

  dim3 gq(NWIN * NPAD / 16, QKVO / 64);
  k_qkv<<<gq, 32, 0, stream>>>(bufA, qkvW, qkv_b, bufB);

  k_attn<<<NWIN * 6, 32, 0, stream>>>(bufB, table, epi, bufA);

  dim3 gp(NWIN * NPAD / 16, CDIM / 64);
  k_proj<<<gp, 32, 0, stream>>>(bufA, projW, proj_b, x, bufX);

  k_ln2<<<TROWS / 8, 256, 0, stream>>>(bufX, n2w, n2b, bufA);

  dim3 g1(TROWS / 16, HID / 64);
  k_fc1<<<g1, 32, 0, stream>>>(bufA, fc1W, fc1_b, bufB);

  dim3 g2(TROWS / 16, CDIM / 64);
  k_fc2<<<g2, 32, 0, stream>>>(bufB, fc2W, fc2_b, bufX, (float*)d_out);
}
